// PatchLoss_45561013076559
// MI455X (gfx1250) — matrix operand order repaired and checked by reading, not compiled, not run on hardware
//
#include <hip/hip_runtime.h>
#include <hip/hip_bf16.h>

typedef __attribute__((ext_vector_type(16))) _Float16 v16h;
typedef __attribute__((ext_vector_type(8)))  _Float16 v8h;
typedef __attribute__((ext_vector_type(8)))  float    v8f;
typedef __attribute__((ext_vector_type(4)))  float    f4;

#define B_ 16
#define T_ 1024
#define D_ 256
#define SSPLIT 8                       // s-dimension split per t-slab
#define SCHUNK (T_ / SSPLIT)           // 128 s-columns per block
#define NBLOCKS_MAIN (B_ * (T_ / 128) * SSPLIT)   // 16*8*8 = 1024

// ---------------------------------------------------------------------------
// Prep: z (f32, [B*T, D]) -> zh (f16) and z2[row] = sum_d z^2
// one block (256 threads) per row
// ---------------------------------------------------------------------------
__global__ void __launch_bounds__(256)
patchloss_prep(const float* __restrict__ z,
               _Float16* __restrict__ zh,
               float* __restrict__ z2) {
    const int row = blockIdx.x;          // 0 .. B*T-1
    const int tid = threadIdx.x;         // 0 .. 255
    const float v = z[(size_t)row * D_ + tid];
    zh[(size_t)row * D_ + tid] = (_Float16)v;

    float sq = v * v;
    #pragma unroll
    for (int off = 16; off > 0; off >>= 1)
        sq += __shfl_xor(sq, off, 32);

    __shared__ float smem[8];
    const int lane = tid & 31, wv = tid >> 5;
    if (lane == 0) smem[wv] = sq;
    __syncthreads();
    if (tid == 0) {
        float s = 0.f;
        #pragma unroll
        for (int i = 0; i < 8; ++i) s += smem[i];
        z2[row] = s;
    }
}

// ---------------------------------------------------------------------------
// Main fused kernel. Grid = B * (T/128) * SSPLIT blocks of 256 threads.
// Each wave owns a 16-row t-tile and a 128-column s-range:
//   - A panel (16x256 f16) resident in VGPRs, loaded once
//   - per s-tile: 8x v_wmma_f32_16x16x32_f16 build Gram tile G
//   - w tile computed in registers in the exact WMMA C/D lane layout
//   - fused reduction of w * (z2[t] + z2[s] - 2G)
// gt_dT is streamed once with non-temporal hints (zero reuse).
// ---------------------------------------------------------------------------
__global__ void __launch_bounds__(256)
patchloss_main(const _Float16* __restrict__ zh,
               const float* __restrict__ z2,
               const float* __restrict__ gt,      // [B,T,T,4]
               const float* __restrict__ sigma,   // [4]
               float* __restrict__ partials) {
    const int lane = threadIdx.x & 31;
    const int wv   = threadIdx.x >> 5;           // 0..7
    const int b    = blockIdx.x >> 6;            // / (8*SSPLIT)
    const int rem  = blockIdx.x & 63;
    const int tblk = rem >> 3;                   // 0..7
    const int sblk = rem & 7;                    // 0..SSPLIT-1
    const int t0   = tblk * 128 + wv * 16;
    const int sbeg = sblk * SCHUNK;
    const int g    = lane >> 4;                  // lane group (0/1)
    const int l16  = lane & 15;

    // inv(2*sigma^2)
    const f4 sg = *(const f4*)sigma;
    const float c0 = 1.f / (2.f * sg.x * sg.x);
    const float c1 = 1.f / (2.f * sg.y * sg.y);
    const float c2 = 1.f / (2.f * sg.z * sg.z);
    const float c3 = 1.f / (2.f * sg.w * sg.w);

    // Load A panel: 8 K-chunks of the 16x256 t-tile, per ISA 16-bit A layout:
    // lane group g: elems 0..7 = K=d0+8g.., elems 8..15 = K=d0+16+8g..
    v16h A[8];
    const _Float16* zrowA = zh + ((size_t)(b * T_ + t0 + l16)) * D_;
    #pragma unroll
    for (int c = 0; c < 8; ++c) {
        const int d0 = c * 32;
        v8h lo = *(const v8h*)(zrowA + d0 + 8 * g);
        v8h hi = *(const v8h*)(zrowA + d0 + 16 + 8 * g);
        v16h a;
        #pragma unroll
        for (int i = 0; i < 8; ++i) { a[i] = lo[i]; a[8 + i] = hi[i]; }
        A[c] = a;
    }

    // z2 for this lane's 8 t-rows (C/D layout: M = r + 8*g)
    float z2t[8];
    #pragma unroll
    for (int r = 0; r < 8; ++r)
        z2t[r] = z2[b * T_ + t0 + 8 * g + r];

    const float* z2b = z2 + b * T_;
    float acc = 0.f;

    for (int s0 = sbeg; s0 < sbeg + SCHUNK; s0 += 16) {
        // B panel rows (z for s-tile) — B layout: N=l16, elems 0..7 = K=d0+8g.., elems 8..15 = K=d0+16+8g..
        const _Float16* zrowB = zh + ((size_t)(b * T_ + s0 + l16)) * D_;
        v8f gacc = {};
        #pragma unroll
        for (int c = 0; c < 8; ++c) {
            v8h blo = *(const v8h*)(zrowB + c * 32 + 8 * g);
            v8h bhi = *(const v8h*)(zrowB + c * 32 + 16 + 8 * g);
            v16h bb;
            #pragma unroll
            for (int i = 0; i < 8; ++i) { bb[i] = blo[i]; bb[8 + i] = bhi[i]; }
            gacc = __builtin_amdgcn_wmma_f32_16x16x32_f16(
                false, A[c], false, bb, (short)0, gacc, false, false);
        }

        const float z2s = z2b[s0 + l16];
        // gt_dT tile: lane covers (M = 8g + r, N = l16), 4 contiguous floats.
        // Streamed exactly once -> non-temporal loads keep L2 for z.
        const f4* gtbase = (const f4*)
            (gt + ((((size_t)b * T_ + (t0 + 8 * g)) * T_) + s0 + l16) * 4);
        #pragma unroll
        for (int r = 0; r < 8; ++r) {
            const f4 d = __builtin_nontemporal_load(gtbase + (size_t)r * T_);
            const float ker = d.x * d.x * c0 + d.y * d.y * c1 +
                              d.z * d.z * c2 + d.w * d.w * c3;
            const float w = __expf(-ker);
            acc += w * (z2t[r] + z2s - 2.0f * gacc[r]);
        }
    }

    // wave reduction (wave32)
    #pragma unroll
    for (int off = 16; off > 0; off >>= 1)
        acc += __shfl_xor(acc, off, 32);

    __shared__ float smem[8];
    if (lane == 0) smem[wv] = acc;
    __syncthreads();
    if (threadIdx.x == 0) {
        float s = 0.f;
        #pragma unroll
        for (int i = 0; i < 8; ++i) s += smem[i];
        partials[blockIdx.x] = s;
    }
}

// ---------------------------------------------------------------------------
// Deterministic final reduction: fixed-order over NBLOCKS_MAIN partials
// ---------------------------------------------------------------------------
__global__ void __launch_bounds__(256)
patchloss_reduce(const float* __restrict__ partials, float* __restrict__ out) {
    __shared__ float sm[256];
    const int tid = threadIdx.x;
    float s = 0.f;
    #pragma unroll
    for (int i = 0; i < NBLOCKS_MAIN / 256; ++i)
        s += partials[tid + i * 256];
    sm[tid] = s;
    __syncthreads();
    #pragma unroll
    for (int off = 128; off > 0; off >>= 1) {
        if (tid < off) sm[tid] += sm[tid + off];
        __syncthreads();
    }
    if (tid == 0)
        out[0] = sm[0] * (1.0f / ((float)B_ * (float)T_ * (float)T_));
}

extern "C" void kernel_launch(void* const* d_in, const int* in_sizes, int n_in,
                              void* d_out, int out_size, void* d_ws, size_t ws_size,
                              hipStream_t stream) {
    (void)in_sizes; (void)n_in; (void)out_size; (void)ws_size;
    const float* z     = (const float*)d_in[0];   // [B,T,D] f32
    const float* gt    = (const float*)d_in[1];   // [B,T,T,4] f32
    const float* sigma = (const float*)d_in[2];   // [4] f32
    float* out = (float*)d_out;

    char* ws = (char*)d_ws;
    _Float16* zh = (_Float16*)ws;                               // 8 MB
    float* z2 = (float*)(ws + (size_t)B_ * T_ * D_ * sizeof(_Float16));
    float* partials = z2 + (size_t)B_ * T_;                     // 1024 floats

    patchloss_prep<<<B_ * T_, 256, 0, stream>>>(z, zh, z2);
    patchloss_main<<<NBLOCKS_MAIN, 256, 0, stream>>>(zh, z2, gt, sigma, partials);
    patchloss_reduce<<<1, 256, 0, stream>>>(partials, out);
}
